// ContextualNeuronPool_82901458747578
// MI455X (gfx1250) — matrix operand order repaired and checked by reading, not compiled, not run on hardware
//
#include <hip/hip_runtime.h>
#include <hip/hip_bf16.h>
#include <math.h>

#define POOL    512
#define DMODEL  1024
#define DFF     4096
#define MDIM    64
#define KSEL    8
#define NTOK    4096   // B*S = 2*2048

typedef __bf16  v16bf __attribute__((ext_vector_type(16)));
typedef float   v8f   __attribute__((ext_vector_type(8)));
typedef float   f4v   __attribute__((ext_vector_type(4)));
typedef unsigned int u4v __attribute__((ext_vector_type(4)));
typedef unsigned short us4v __attribute__((ext_vector_type(4)));

static __device__ __forceinline__ unsigned short f32_to_bf16(float f) {
    union { float f; unsigned u; } in; in.f = f;
    unsigned u = in.u;
    if ((u & 0x7fffffffu) > 0x7f800000u)            // NaN: quiet it
        return (unsigned short)((u >> 16) | 0x0040u);
    unsigned r = u + 0x7fffu + ((u >> 16) & 1u);    // round-to-nearest-even
    return (unsigned short)(r >> 16);
}

// ---------------------------------------------------------------------------
// Kernel 1: one-time f32 -> bf16 conversion of w2_w (1024 x 4096)
// ---------------------------------------------------------------------------
__global__ __launch_bounds__(256) void cvt_w2_bf16(const float* __restrict__ src,
                                                   unsigned short* __restrict__ dst) {
    int i = blockIdx.x * 256 + threadIdx.x;          // each thread: 4 elements
    f4v v = ((const f4v*)src)[i];
    us4v o;
    o.x = f32_to_bf16(v.x); o.y = f32_to_bf16(v.y);
    o.z = f32_to_bf16(v.z); o.w = f32_to_bf16(v.w);
    ((us4v*)dst)[i] = o;
}

// ---------------------------------------------------------------------------
// Kernel 2: per-token fused stage.
//   - softmax(pattern_weights)
//   - sel_mod_k = x @ cm_w[idx_k block]^T + cm_b   (only the K=8 selected blocks)
//   - wmod = sum_k w_k * sel_mod_k                 (fold combine before adj_proj)
//   - combined = sum_k w_k * bp[idx_k] + wmod @ adj_proj
//   - activated = gelu_exact(combined)  -> bf16 to workspace
// One block (256 thr = 8 waves) per token; wave k handles selected index k.
// ---------------------------------------------------------------------------
__global__ __launch_bounds__(256) void fused_token(
    const float* __restrict__ x,
    const int*   __restrict__ sel,
    const float* __restrict__ pw,
    const float* __restrict__ bp,
    const float* __restrict__ cm_w,
    const float* __restrict__ cm_b,
    const float* __restrict__ adj,
    unsigned short* __restrict__ act_bf16)
{
    __shared__ float x_s[DMODEL];
    __shared__ float w_s[KSEL];
    __shared__ int   idx_s[KSEL];
    __shared__ float acc_s[KSEL][MDIM];
    __shared__ float wmod_s[MDIM];

    const int t   = blockIdx.x;
    const int tid = threadIdx.x;

    // stage x[t] (4KB) into LDS, vectorized
    ((f4v*)x_s)[tid] = ((const f4v*)(x + (size_t)t * DMODEL))[tid];

    if (tid < KSEL) idx_s[tid] = sel[t * KSEL + tid];
    if (tid == 0) {
        float w[KSEL]; float mx = -1e30f;
        for (int k = 0; k < KSEL; ++k) { w[k] = pw[t * KSEL + k]; mx = fmaxf(mx, w[k]); }
        float s = 0.f;
        for (int k = 0; k < KSEL; ++k) { w[k] = __expf(w[k] - mx); s += w[k]; }
        float inv = 1.f / s;
        for (int k = 0; k < KSEL; ++k) w_s[k] = w[k] * inv;
    }
    __syncthreads();

    // wave k computes 64 dots of length 1024 for its selected index
    const int k  = tid >> 5;     // wave id == selected-index slot
    const int ln = tid & 31;
    {
        const int   nidx  = idx_s[k];
        const float wk    = w_s[k];
        const float* wrow0 = cm_w + ((size_t)nidx * MDIM) * DMODEL;
        for (int m = 0; m < MDIM; ++m) {
            const float* wr = wrow0 + (size_t)m * DMODEL;
            float d = 0.f;
            #pragma unroll 8
            for (int j = ln; j < DMODEL; j += 32) d += x_s[j] * wr[j];
            #pragma unroll
            for (int off = 16; off > 0; off >>= 1) d += __shfl_xor(d, off, 32);
            if (ln == 0) acc_s[k][m] = wk * (d + cm_b[nidx * MDIM + m]);
        }
    }
    __syncthreads();
    if (tid < MDIM) {
        float s = 0.f;
        #pragma unroll
        for (int kk = 0; kk < KSEL; ++kk) s += acc_s[kk][tid];
        wmod_s[tid] = s;
    }
    __syncthreads();

    // combined + exact-erf gelu + bf16 store (coalesced across tid)
    const size_t obase = (size_t)t * DFF;
    for (int i = 0; i < DFF / 256; ++i) {
        const int f = tid + 256 * i;
        float a = 0.f;
        #pragma unroll
        for (int kk = 0; kk < KSEL; ++kk)
            a += w_s[kk] * bp[(size_t)idx_s[kk] * DFF + f];
        #pragma unroll 8
        for (int m = 0; m < MDIM; ++m)
            a += wmod_s[m] * adj[(size_t)m * DFF + f];
        float g = 0.5f * a * (1.f + erff(a * 0.70710678118654752440f));
        act_bf16[obase + f] = f32_to_bf16(g);
    }
}

// ---------------------------------------------------------------------------
// Kernel 3: out[4096,1024] = act[4096,4096](bf16) x w2_w^T (bf16) + bias
// Register-blocked 2x2 WMMA tiles per wave (32x32 output), 4 independent
// accumulation chains per K-step, 16 FLOP/byte from L2.
// Per-lane operand layout per CDNA5 ISA 7.12.2 (16-bit A 16x32 / B 32x16):
//   A: lanes 0-15 row=lane, K {0..7,16..23}; lanes 16-31 row=lane-16, K {8..15,24..31}
//   B: lanes 0-15 col=lane, K {0..7,16..23}; lanes 16-31 col=lane-16, K {8..15,24..31}
//   C/D f32: VGPR r: lanes 0-15 -> (M=r, N=lane); lanes 16-31 -> (M=r+8, N=lane-16)
// ---------------------------------------------------------------------------
__global__ __launch_bounds__(256) void wmma_gemm(
    const unsigned short* __restrict__ A,   // [NTOK, DFF] bf16
    const unsigned short* __restrict__ Wt,  // [DMODEL, DFF] bf16
    const float* __restrict__ bias,         // [DMODEL]
    float* __restrict__ out)                // [NTOK, DMODEL]
{
    const int wv = threadIdx.x >> 5;
    const int ln = threadIdx.x & 31;
    const int g  = blockIdx.x * 8 + wv;     // global wave-supertile id
    const int mt = g >> 5;                  // / (DMODEL/32 = 32)
    const int nt = g & 31;
    const int row0 = mt * 32;
    const int n0   = nt * 32;

    const int lr = ln & 15;
    const int hi = ln >> 4;

    const unsigned short* Arow0 = A  + (size_t)(row0 + lr) * DFF + (hi ? 8 : 0);
    const unsigned short* Arow1 = Arow0 + (size_t)16 * DFF;
    const unsigned short* Brow0 = Wt + (size_t)(n0  + lr) * DFF + (hi ? 8 : 0);
    const unsigned short* Brow1 = Brow0 + (size_t)16 * DFF;

    v8f acc00 = {0.f,0.f,0.f,0.f,0.f,0.f,0.f,0.f};
    v8f acc01 = acc00, acc10 = acc00, acc11 = acc00;

    #pragma unroll 2
    for (int kb = 0; kb < DFF; kb += 32) {
        union { u4v u[2]; v16bf v; } a0, a1, b0, b1;
        const u4v* p;
        p = (const u4v*)(Arow0 + kb); a0.u[0] = p[0]; a0.u[1] = p[2];
        p = (const u4v*)(Arow1 + kb); a1.u[0] = p[0]; a1.u[1] = p[2];
        p = (const u4v*)(Brow0 + kb); b0.u[0] = p[0]; b0.u[1] = p[2];
        p = (const u4v*)(Brow1 + kb); b1.u[0] = p[0]; b1.u[1] = p[2];
        __builtin_prefetch(Arow0 + kb + 512, 0, 0);
        __builtin_prefetch(Arow1 + kb + 512, 0, 0);
        acc00 = __builtin_amdgcn_wmma_f32_16x16x32_bf16(
                    false, a0.v, false, b0.v, (short)0, acc00, false, false);
        acc01 = __builtin_amdgcn_wmma_f32_16x16x32_bf16(
                    false, a0.v, false, b1.v, (short)0, acc01, false, false);
        acc10 = __builtin_amdgcn_wmma_f32_16x16x32_bf16(
                    false, a1.v, false, b0.v, (short)0, acc10, false, false);
        acc11 = __builtin_amdgcn_wmma_f32_16x16x32_bf16(
                    false, a1.v, false, b1.v, (short)0, acc11, false, false);
    }

    const float bn0 = bias[n0 + lr];
    const float bn1 = bias[n0 + 16 + lr];
    float* o00 = out + (size_t)(row0 + hi * 8) * DMODEL + n0 + lr;
    #pragma unroll
    for (int r = 0; r < 8; ++r) {
        float* orow = o00 + (size_t)r * DMODEL;
        orow[0]                        = acc00[r] + bn0;   // (row0 + M,  n0 + N)
        orow[16]                       = acc01[r] + bn1;   // (row0 + M,  n0+16+N)
        orow[(size_t)16 * DMODEL]      = acc10[r] + bn0;   // (row0+16+M, n0 + N)
        orow[(size_t)16 * DMODEL + 16] = acc11[r] + bn1;   // (row0+16+M, n0+16+N)
    }
}

// ---------------------------------------------------------------------------
extern "C" void kernel_launch(void* const* d_in, const int* in_sizes, int n_in,
                              void* d_out, int out_size, void* d_ws, size_t ws_size,
                              hipStream_t stream) {
    const float* x    = (const float*)d_in[0];
    const int*   sel  = (const int*)  d_in[1];
    const float* pw   = (const float*)d_in[2];
    const float* bp   = (const float*)d_in[3];
    const float* cm_w = (const float*)d_in[4];
    const float* cm_b = (const float*)d_in[5];
    const float* adj  = (const float*)d_in[6];
    const float* w2w  = (const float*)d_in[7];
    const float* w2b  = (const float*)d_in[8];
    float* out = (float*)d_out;

    unsigned short* act  = (unsigned short*)d_ws;                 // 32 MB
    unsigned short* w2bf = act + (size_t)NTOK * DFF;              // +8 MB

    // 1024*4096 elems / (256 thr * 4 per thr) = 4096 blocks
    cvt_w2_bf16<<<(DMODEL * DFF) / (256 * 4), 256, 0, stream>>>(w2w, w2bf);
    fused_token<<<NTOK, 256, 0, stream>>>(x, sel, pw, bp, cm_w, cm_b, adj, act);
    // (4096/32)*(1024/32) = 4096 wave-supertiles / 8 waves per block = 512 blocks
    wmma_gemm<<<(NTOK / 32) * (DMODEL / 32) / 8, 256, 0, stream>>>(act, w2bf, w2b, out);
}
